// MHCA_18940805775521
// MI455X (gfx1250) — matrix operand order repaired and checked by reading, not compiled, not run on hardware
//
#include <hip/hip_runtime.h>

// ---------------------------------------------------------------------------
// MHCA forward for MI455X (gfx1250), wave32, v_wmma_f32_16x16x32_f16 +
// global_load_async_to_lds_b128 double-buffered staging (ASYNCcnt).
// B=4, Nq=Nk=1024, D=512, H=8, dh=64. f16 operands, f32 accumulate.
// d_out = [out (B,Nq,D) f32][w (B,H,Nq,Nk) f32].
// ---------------------------------------------------------------------------

#define BD    4
#define NQ    1024
#define NK    1024
#define DM    512
#define HEADS 8
#define DH    64

typedef __attribute__((ext_vector_type(16))) _Float16 v16h;
typedef __attribute__((ext_vector_type(8)))  float    v8f;

// CDNA5 WMMA 16-bit A-matrix 16x32 layout (ISA 7.12.2):
// lane<16: m=lane, j<4 -> K=2j, j>=4 -> K=8+2j; lane>=16: K offset +8.
__device__ __forceinline__ int a_kbase(int j, int lane) {
    return (j < 4 ? 2 * j : 8 + 2 * j) + ((lane & 16) ? 8 : 0);
}
// 16-bit B-matrix 32x16 layout: lane<16: n=lane, j<4 -> K=2j, j>=4 -> K=8+2j; lane>=16: K offset +8.
__device__ __forceinline__ int b_kbase(int j, int lane) {
    return (j < 4 ? 2 * j : 8 + 2 * j) + ((lane & 16) ? 8 : 0);
}

__device__ __forceinline__ v8f wmma_f16(v16h a, v16h b, v8f c) {
    return __builtin_amdgcn_wmma_f32_16x16x32_f16(false, a, false, b,
                                                  (short)0, c, false, false);
}

// CDNA5 async copy global -> LDS (16B per lane), tracked by ASYNCcnt.
__device__ __forceinline__ void async_b128(void* lds_ptr, const _Float16* g) {
    unsigned int off = (unsigned int)(unsigned long long)lds_ptr; // LDS offset
    asm volatile("global_load_async_to_lds_b128 %0, %1, off"
                 :: "v"(off), "v"(g) : "memory");
}
__device__ __forceinline__ void wait_async0() {
    asm volatile("s_wait_asynccnt 0" ::: "memory");
}

// ---------------------------------------------------------------------------
// Prep 1: f32 -> f16 elementwise (qx, kx).
// ---------------------------------------------------------------------------
__global__ __launch_bounds__(256) void mhca_cvt_kernel(
    const float* __restrict__ src, _Float16* __restrict__ dst, int n)
{
    int i = (blockIdx.x * 256 + threadIdx.x) * 4;
    if (i + 3 < n) {
        float4 v = *(const float4*)(src + i);
        dst[i + 0] = (_Float16)v.x;
        dst[i + 1] = (_Float16)v.y;
        dst[i + 2] = (_Float16)v.z;
        dst[i + 3] = (_Float16)v.w;
    }
}

// ---------------------------------------------------------------------------
// Prep 2: W[k][c] f32 -> Wt[c][k] f16 (transpose+convert), LDS 32x33 tile.
// blockIdx.z selects Wq/Wk/Wv/Wo.
// ---------------------------------------------------------------------------
__global__ __launch_bounds__(256) void mhca_wt_kernel(
    const float* __restrict__ W0, const float* __restrict__ W1,
    const float* __restrict__ W2, const float* __restrict__ W3,
    _Float16* __restrict__ WtBase)
{
    const float* W = (blockIdx.z == 0) ? W0 : (blockIdx.z == 1) ? W1
                   : (blockIdx.z == 2) ? W2 : W3;
    _Float16* Wt = WtBase + (size_t)blockIdx.z * DM * DM;

    __shared__ float tile[32][33];
    const int tx = threadIdx.x & 31;      // 32
    const int ty = threadIdx.x >> 5;      // 8
    const int k0 = blockIdx.y * 32;
    const int c0 = blockIdx.x * 32;
#pragma unroll
    for (int i = 0; i < 4; ++i)
        tile[ty + 8 * i][tx] = W[(size_t)(k0 + ty + 8 * i) * DM + c0 + tx];
    __syncthreads();
#pragma unroll
    for (int i = 0; i < 4; ++i)
        Wt[(size_t)(c0 + ty + 8 * i) * DM + k0 + tx] =
            (_Float16)tile[tx][ty + 8 * i];
}

// ---------------------------------------------------------------------------
// Kernel 1: Y = Xh @ W + bias (f16 in, f16 out), Wt pre-transposed [c][k].
// 256 thr = 8 waves; block tile 64x64; wave tile 16x32 (1 A frag, 2 accs).
// A/B tiles staged with async b128 copies, double buffered.
// blockIdx.z: 0 = Q(from qx), 1 = K(from kx), 2 = V(from kx).
// ---------------------------------------------------------------------------
__global__ __launch_bounds__(256) void mhca_proj_kernel(
    const _Float16* __restrict__ Xq, const _Float16* __restrict__ Xk,
    const _Float16* __restrict__ WtAll,
    const float* __restrict__ bq, const float* __restrict__ bk,
    const float* __restrict__ bv,
    _Float16* __restrict__ ws)
{
    const int which = blockIdx.z;
    const _Float16* X  = (which == 0) ? Xq : Xk;
    const _Float16* Wt = WtAll + (size_t)which * DM * DM;
    const float* bias  = (which == 0) ? bq : (which == 1 ? bk : bv);
    _Float16* Y = ws + (size_t)which * ((size_t)BD * NQ * DM);

    __shared__ __align__(16) _Float16 ldsA[2][64][32];  // X rows  [m][k]
    __shared__ __align__(16) _Float16 ldsB[2][64][32];  // Wt rows [n][k]

    const int t    = threadIdx.x;
    const int lane = t & 31;
    const int wave = t >> 5;
    const int wr   = wave >> 1;     // 0..3
    const int wc   = wave & 1;      // 0..1
    const int row0 = blockIdx.x * 64;
    const int col0 = blockIdx.y * 64;

    const int srow = t >> 2;        // 0..63: staged row
    const int sseg = (t & 3) * 8;   // 16B segment within 64B row

    // stage step k0 into buffer `buf`: 2 async b128 per thread
    auto stage = [&](int k0, int buf) {
        async_b128(&ldsA[buf][srow][sseg],
                   X + (size_t)(row0 + srow) * DM + k0 + sseg);
        async_b128(&ldsB[buf][srow][sseg],
                   Wt + (size_t)(col0 + srow) * DM + k0 + sseg);
    };

    v8f acc0 = {}; v8f acc1 = {};
    stage(0, 0);
#pragma unroll
    for (int kt = 0; kt < DM / 32; ++kt) {
        const int buf = kt & 1;
        wait_async0();
        __syncthreads();
        if (kt + 1 < DM / 32) stage((kt + 1) * 32, buf ^ 1);

        v16h af, bf0, bf1;
        const int m = wr * 16 + (lane & 15);
        const int n = wc * 32 + (lane & 15);
#pragma unroll
        for (int j = 0; j < 8; ++j) {
            int ka = a_kbase(j, lane);
            af[2 * j]     = ldsA[buf][m][ka];
            af[2 * j + 1] = ldsA[buf][m][ka + 1];
            int kb = b_kbase(j, lane);
            bf0[2 * j]     = ldsB[buf][n][kb];
            bf0[2 * j + 1] = ldsB[buf][n][kb + 1];
            bf1[2 * j]     = ldsB[buf][n + 16][kb];
            bf1[2 * j + 1] = ldsB[buf][n + 16][kb + 1];
        }
        acc0 = wmma_f16(af, bf0, acc0);
        acc1 = wmma_f16(af, bf1, acc1);
    }

    // epilogue: + bias, store f16 per C/D layout (m = r + 8*(lane>=16))
    const int n   = lane & 15;
    const int gr0 = row0 + wr * 16 + ((lane & 16) ? 8 : 0);
    const int gc0 = col0 + wc * 32 + n;
    const float bb0 = bias[gc0];
    const float bb1 = bias[gc0 + 16];
#pragma unroll
    for (int r = 0; r < 8; ++r) {
        Y[(size_t)(gr0 + r) * DM + gc0]      = (_Float16)(acc0[r] + bb0);
        Y[(size_t)(gr0 + r) * DM + gc0 + 16] = (_Float16)(acc1[r] + bb1);
    }
}

// ---------------------------------------------------------------------------
// Kernel 2: attention for one (b, h, 16 q-rows). 128 thr = 4 waves.
// Phase A: S = QK^T/8 (+mask,+bias) via WMMA into 64KB LDS strip.
// Phase B: row softmax (shuffle reduce), write w once to d_out.
// Phase C: ctx = w @ V via WMMA, f16 to workspace.
// ---------------------------------------------------------------------------
__global__ __launch_bounds__(128) void mhca_attn_kernel(
    const _Float16* __restrict__ Qh, const _Float16* __restrict__ Kh,
    const _Float16* __restrict__ Vh,
    const unsigned char* __restrict__ mask_k,
    const float* __restrict__ log_g,
    float* __restrict__ w_out,
    _Float16* __restrict__ Ctxh)
{
    __shared__ float S[16][NK];   // 64 KB score strip

    const int t    = threadIdx.x;
    const int lane = t & 31;
    const int wave = t >> 5;      // 0..3
    const int q0   = blockIdx.x * 16;
    const int h    = blockIdx.y;
    const int b    = blockIdx.z;

    // Q fragments (rows q0..q0+15, dh=64 -> two K=32 chunks), loaded once.
    v16h qa0, qa1;
    {
        const _Float16* Qbase = Qh + (size_t)(b * NQ + q0) * DM + h * DH;
        const int m = lane & 15;
#pragma unroll
        for (int j = 0; j < 8; ++j) {
            int ka = a_kbase(j, lane);
            qa0[2 * j]     = Qbase[(size_t)m * DM + ka];
            qa0[2 * j + 1] = Qbase[(size_t)m * DM + ka + 1];
            qa1[2 * j]     = Qbase[(size_t)m * DM + 32 + ka];
            qa1[2 * j + 1] = Qbase[(size_t)m * DM + 32 + ka + 1];
        }
    }

    // ---- Phase A: 64 k-tiles of 16, split over 4 waves.
    const float scale = 0.125f;   // 1/sqrt(64)
    for (int kt = wave; kt < NK / 16; kt += 4) {
        const int kc0 = kt * 16;
        const _Float16* Kbase = Kh + (size_t)(b * NK + kc0) * DM + h * DH;
        const int n = lane & 15;
        v16h kb0, kb1;
#pragma unroll
        for (int j = 0; j < 8; ++j) {
            int kd = b_kbase(j, lane);           // B[k=d][n=kcol] = K[kcol][d]
            kb0[2 * j]     = Kbase[(size_t)n * DM + kd];
            kb0[2 * j + 1] = Kbase[(size_t)n * DM + kd + 1];
            kb1[2 * j]     = Kbase[(size_t)n * DM + 32 + kd];
            kb1[2 * j + 1] = Kbase[(size_t)n * DM + 32 + kd + 1];
        }
        v8f acc = {};
        acc = wmma_f16(qa0, kb0, acc);
        acc = wmma_f16(qa1, kb1, acc);

        const int   col = kc0 + n;
        const float gb  = log_g[b * NK + col];
        const bool  mk  = mask_k[b * NK + col] != 0;
        const int   m0  = (lane & 16) ? 8 : 0;
#pragma unroll
        for (int r = 0; r < 8; ++r) {
            float sv = acc[r] * scale;
            sv = mk ? (sv + gb) : -__builtin_inff();
            S[m0 + r][col] = sv;
        }
    }
    __syncthreads();

    // ---- Phase B: softmax, 8 contiguous lanes per row.
    {
        const int r  = t >> 3;    // 0..15
        const int jj = t & 7;
        float mx = -__builtin_inff();
        for (int c = jj; c < NK; c += 8) mx = fmaxf(mx, S[r][c]);
#pragma unroll
        for (int off = 1; off < 8; off <<= 1) mx = fmaxf(mx, __shfl_xor(mx, off, 32));
        float sum = 0.f;
        for (int c = jj; c < NK; c += 8) {
            float e = __expf(S[r][c] - mx);
            S[r][c] = e;
            sum += e;
        }
#pragma unroll
        for (int off = 1; off < 8; off <<= 1) sum += __shfl_xor(sum, off, 32);
        const float inv = 1.0f / sum;
        float* wrow = w_out + ((size_t)(b * HEADS + h) * NQ + q0 + r) * NK;
        for (int c = jj; c < NK; c += 8) {
            float p = S[r][c] * inv;
            S[r][c] = p;       // keep normalized probs for phase C
            wrow[c] = p;       // single HBM touch of the 128MB w slab
        }
    }
    __syncthreads();

    // ---- Phase C: ctx = w @ V. Wave owns dh cols [wave*16, wave*16+16).
    v8f cacc = {};
    const int n = lane & 15;
    const int m = lane & 15;
    for (int kk = 0; kk < NK; kk += 32) {
        const _Float16* Vbase =
            Vh + (size_t)(b * NK + kk) * DM + h * DH + wave * 16;
        v16h af, bf;
#pragma unroll
        for (int j = 0; j < 8; ++j) {
            int ka = a_kbase(j, lane);
            af[2 * j]     = (_Float16)S[m][kk + ka];
            af[2 * j + 1] = (_Float16)S[m][kk + ka + 1];
            int kd = b_kbase(j, lane);           // B[k=krow][n=dcol]
            bf[2 * j]     = Vbase[(size_t)kd * DM + n];
            bf[2 * j + 1] = Vbase[(size_t)(kd + 1) * DM + n];
        }
        cacc = wmma_f16(af, bf, cacc);
    }
    _Float16* Cbase = Ctxh + (size_t)(b * NQ + q0) * DM + h * DH + wave * 16;
    const int m0 = (lane & 16) ? 8 : 0;
#pragma unroll
    for (int r = 0; r < 8; ++r)
        Cbase[(size_t)(m0 + r) * DM + n] = (_Float16)cacc[r];
}

// ---------------------------------------------------------------------------
// Kernel 3: x = ctx @ Wo + bo; out = LayerNorm(x + qx). Wot pre-transposed.
// 256 thr, 16-row strip; wave owns 64 cols (4 accs sharing the A frag).
// ---------------------------------------------------------------------------
__global__ __launch_bounds__(256) void mhca_out_ln_kernel(
    const _Float16* __restrict__ Ctxh, const _Float16* __restrict__ Wot,
    const float* __restrict__ bo, const float* __restrict__ qx,
    const float* __restrict__ ln_g, const float* __restrict__ ln_b,
    float* __restrict__ out)
{
    __shared__ float xs[16][DM];   // 32 KB

    const int t    = threadIdx.x;
    const int lane = t & 31;
    const int wave = t >> 5;
    const int row0 = blockIdx.x * 16;

    v8f acc[4] = {{}, {}, {}, {}};
    const int m = lane & 15;
    const int n = lane & 15;
#pragma unroll 4
    for (int k0 = 0; k0 < DM; k0 += 32) {
        v16h af;
        const _Float16* Abase = Ctxh + (size_t)row0 * DM + k0;
#pragma unroll
        for (int j = 0; j < 8; ++j) {
            int ka = a_kbase(j, lane);
            af[2 * j]     = Abase[(size_t)m * DM + ka];
            af[2 * j + 1] = Abase[(size_t)m * DM + ka + 1];
        }
#pragma unroll
        for (int cg = 0; cg < 4; ++cg) {
            // Wot[c][k]: contiguous in k -> per-lane b128 pairs
            const _Float16* Wbase =
                Wot + (size_t)(wave * 64 + cg * 16 + n) * DM + k0;
            v16h bf;
#pragma unroll
            for (int j = 0; j < 8; ++j) {
                int kd = b_kbase(j, lane);
                bf[2 * j]     = Wbase[kd];
                bf[2 * j + 1] = Wbase[kd + 1];
            }
            acc[cg] = wmma_f16(af, bf, acc[cg]);
        }
    }
    const int m0 = (lane & 16) ? 8 : 0;
#pragma unroll
    for (int cg = 0; cg < 4; ++cg) {
        int c = wave * 64 + cg * 16 + n;
        float bb = bo[c];
#pragma unroll
        for (int r = 0; r < 8; ++r) xs[m0 + r][c] = acc[cg][r] + bb;
    }
    __syncthreads();

    // residual + LayerNorm: 16 contiguous lanes per row
    const int r  = t >> 4;   // 0..15
    const int jj = t & 15;
    const float* qrow = qx + (size_t)(row0 + r) * DM;
    float sum = 0.f, sumsq = 0.f;
    for (int c = jj; c < DM; c += 16) {
        float y = xs[r][c] + qrow[c];
        sum += y; sumsq += y * y;
    }
#pragma unroll
    for (int off = 1; off < 16; off <<= 1) {
        sum   += __shfl_xor(sum,   off, 32);
        sumsq += __shfl_xor(sumsq, off, 32);
    }
    const float mu   = sum * (1.0f / DM);
    const float var  = sumsq * (1.0f / DM) - mu * mu;
    const float rstd = rsqrtf(var + 1e-5f);
    float* orow = out + (size_t)(row0 + r) * DM;
    for (int c = jj; c < DM; c += 16) {
        float y = xs[r][c] + qrow[c];
        orow[c] = (y - mu) * rstd * ln_g[c] + ln_b[c];
    }
}

// ---------------------------------------------------------------------------
extern "C" void kernel_launch(void* const* d_in, const int* in_sizes, int n_in,
                              void* d_out, int out_size, void* d_ws, size_t ws_size,
                              hipStream_t stream) {
    const float* qx   = (const float*)d_in[0];
    const float* kx   = (const float*)d_in[1];
    // d_in[2] = mask_q (unused by the reference math)
    const unsigned char* mask_k = (const unsigned char*)d_in[3];
    const float* log_g = (const float*)d_in[4];
    const float* Wq = (const float*)d_in[5];
    const float* bq = (const float*)d_in[6];
    const float* Wk = (const float*)d_in[7];
    const float* bk = (const float*)d_in[8];
    const float* Wv = (const float*)d_in[9];
    const float* bv = (const float*)d_in[10];
    const float* Wo = (const float*)d_in[11];
    const float* bo = (const float*)d_in[12];
    const float* ln_g = (const float*)d_in[13];
    const float* ln_b = (const float*)d_in[14];

    const size_t NE = (size_t)BD * NQ * DM;    // 2,097,152 per activation
    const size_t WE = (size_t)DM * DM;         //   262,144 per weight
    _Float16* ws   = (_Float16*)d_ws;
    _Float16* Qh   = ws;                       // f16 activations
    _Float16* Kh   = ws + NE;
    _Float16* Vh   = ws + 2 * NE;
    _Float16* Ctxh = ws + 3 * NE;
    _Float16* Xqh  = ws + 4 * NE;              // converted inputs
    _Float16* Xkh  = ws + 5 * NE;
    _Float16* WtA  = ws + 6 * NE;              // Wq^T|Wk^T|Wv^T|Wo^T f16
    _Float16* Wot  = WtA + 3 * WE;
    float* out   = (float*)d_out;
    float* w_out = out + NE;                   // w slab after (B,Nq,D) out

    // 0) precision prep: f32->f16 inputs, transposed f16 weights
    mhca_cvt_kernel<<<dim3((int)(NE / 4 / 256)), 256, 0, stream>>>(qx, Xqh, (int)NE);
    mhca_cvt_kernel<<<dim3((int)(NE / 4 / 256)), 256, 0, stream>>>(kx, Xkh, (int)NE);
    mhca_wt_kernel<<<dim3(DM / 32, DM / 32, 4), 256, 0, stream>>>(Wq, Wk, Wv, Wo, WtA);
    // 1) Q/K/V projections (async double-buffered LDS staging)
    mhca_proj_kernel<<<dim3((BD * NQ) / 64, DM / 64, 3), 256, 0, stream>>>(
        Xqh, Xkh, WtA, bq, bk, bv, ws);
    // 2) attention: one block per (b, h, 16 q-rows)
    mhca_attn_kernel<<<dim3(NQ / 16, HEADS, BD), 128, 0, stream>>>(
        Qh, Kh, Vh, mask_k, log_g, w_out, Ctxh);
    // 3) output projection + residual + LayerNorm
    mhca_out_ln_kernel<<<dim3((BD * NQ) / 16), 256, 0, stream>>>(
        Ctxh, Wot, bo, qx, ln_g, ln_b, out);
}
